// StatePHGN_66924180406882
// MI455X (gfx1250) — matrix operand order repaired and checked by reading, not compiled, not run on hardware
//
#include <hip/hip_runtime.h>

// ---------------------------------------------------------------------------
// CDNA5 (gfx1250, wave32) fused Port-Hamiltonian RK4 step.
// GEMMs via v_wmma_f32_16x16x32_bf16; weights staged in LDS as bf16.
// ---------------------------------------------------------------------------

typedef __attribute__((ext_vector_type(16))) __bf16       bf16x16;
typedef __attribute__((ext_vector_type(8)))  float        f32x8;
typedef __attribute__((ext_vector_type(4)))  unsigned int u32x4;

union Frag {
    unsigned int u[8];
    u32x4        q[2];
    bf16x16      v;
};

#define HID   256
#define WSTR  264                 // padded ushort row stride (256 + 8) -> 132 dwords/row

// dynamic-LDS byte offsets (all 16B aligned)
#define OFF_W     0               // weights: 256*264*2           = 135168
#define OFF_H1    135168          // act A-staging: 8*16*264*2    = 67584
#define OFF_D2    202752          // grad A-staging:              = 67584
#define OFF_W1S   270336          // layer-1 W+b as float4[256]   = 4096
#define OFF_W3S   274432          // W3 row f32[256]              = 1024
#define OFF_B2S   275456          // b2 f32[256]                  = 1024
#define OFF_ST    276480          // per-wave state 5*[8][16] f32 = 2560
#define SMEM_BYTES 279040

__device__ __forceinline__ unsigned short f2bf(float f) {
    unsigned int x = __float_as_uint(f);
    x += 0x7FFFu + ((x >> 16) & 1u);          // round-to-nearest-even
    return (unsigned short)(x >> 16);
}
__device__ __forceinline__ float softplus_f(float x) {
    float e = __expf(-fabsf(x));
    float r = __logf(1.0f + e);
    return x > 0.0f ? x + r : r;              // log(1+e^x) = max(x,0)+log(1+e^-|x|)
}
__device__ __forceinline__ float sigmoid_f(float x) {
    return 1.0f / (1.0f + __expf(-x));
}

__device__ __forceinline__ f32x8 wmma_bf16(bf16x16 a, bf16x16 b, f32x8 c) {
    // (neg_a, A, neg_b, B, c_mod, C, reuse_a, reuse_b)
    return __builtin_amdgcn_wmma_f32_16x16x32_bf16(false, a, false, b,
                                                   (short)0, c, false, false);
}

// A operand (16x32 bf16): M = lane&15, K = hf*16 + 0..15 (contiguous; B uses same order). Rows 16B aligned.
__device__ __forceinline__ bf16x16 load_act_frag(const unsigned short* act,
                                                 int lw, int hf, int kt) {
    const u32x4* p = (const u32x4*)(act + lw * WSTR);
    Frag f;
    f.q[0] = p[kt * 4 + hf * 2];      // K = kb + hf*16 + {0..7}   (pairs t=0..3)
    f.q[1] = p[kt * 4 + hf * 2 + 1];  // K = kb + hf*16 + 8 + ... (pairs t=4..7)
    return f.v;
}
// B operand, forward: B[k][n] = W[n][k]  -> contiguous dwords within row n
__device__ __forceinline__ bf16x16 load_wfwd_frag(const unsigned short* W,
                                                  int lw, int hf, int nt, int kt) {
    const u32x4* p = (const u32x4*)(W + (nt * 16 + lw) * WSTR);
    Frag f;
    f.q[0] = p[kt * 4 + hf * 2];
    f.q[1] = p[kt * 4 + hf * 2 + 1];
    return f.v;
}
// B operand, backward: B[n][j] = W[n][j] -> column gather, pack 2 rows per dword
__device__ __forceinline__ bf16x16 load_wbwd_frag(const unsigned short* W,
                                                  int lw, int hf, int jt, int kt) {
    int j  = jt * 16 + lw;
    int r0 = kt * 32 + hf * 16;
    Frag f;
#pragma unroll
    for (int t = 0; t < 8; ++t) {
        unsigned int lo = W[(r0 + 2 * t    ) * WSTR + j];
        unsigned int hi = W[(r0 + 2 * t + 1) * WSTR + j];
        f.u[t] = lo | (hi << 16);
    }
    return f.v;
}

__global__ __launch_bounds__(256, 1)
void phgnn_rk4(const float* __restrict__ q,   const float* __restrict__ p,
               const float* __restrict__ u,
               const float* __restrict__ kW1, const float* __restrict__ kb1,
               const float* __restrict__ kW2, const float* __restrict__ kb2,
               const float* __restrict__ kW3,
               const float* __restrict__ pW1, const float* __restrict__ pb1,
               const float* __restrict__ pW2, const float* __restrict__ pb2,
               const float* __restrict__ pW3,
               const float* __restrict__ Amat, const float* __restrict__ Lp,
               const float* __restrict__ bc,
               float* __restrict__ out, int n) {
    extern __shared__ char smem[];
    unsigned short* Wl  = (unsigned short*)(smem + OFF_W);
    unsigned short* H1  = (unsigned short*)(smem + OFF_H1);
    unsigned short* D2  = (unsigned short*)(smem + OFF_D2);
    float4* w1s = (float4*)(smem + OFF_W1S);
    float*  w3s = (float*)(smem + OFF_W3S);
    float*  b2s = (float*)(smem + OFF_B2S);
    float*  st  = (float*)(smem + OFF_ST);
    float* x0a = st;            // sin(q)   [8][16]
    float* x1a = st + 128;      // cos(q)
    float* x2a = st + 256;      // p
    float* gqa = st + 384;      // dH/dq accum
    float* gpa = st + 512;      // dH/dp accum

    const int tid  = threadIdx.x;
    const int lane = tid & 31;
    const int wave = tid >> 5;
    const int lw   = lane & 15;
    const int hf   = lane >> 4;
    const bool owner = (hf == 0);
    const int row = (int)blockIdx.x * 128 + wave * 16 + lw;

    // M = (A - A^T) - L L^T, L = tril(Lp,-1) + diag(softplus(diag(Lp)))   (2x2)
    float jj  = Amat[1] - Amat[2];
    float L00 = softplus_f(Lp[0]);
    float L10 = Lp[2];
    float L11 = softplus_f(Lp[3]);
    float M00 = -(L00 * L00);
    float M01 = jj - L00 * L10;
    float M10 = -jj - L00 * L10;
    float M11 = -(L10 * L10 + L11 * L11);
    float bcv = bc[0];

    float q0 = 0.f, p0 = 0.f, Bu = 0.f;
    if (owner && row < n) { q0 = q[row]; p0 = p[row]; Bu = u[row] * bcv; }

    const float* W2g[2] = {kW2, pW2};
    const float* W1g[2] = {kW1, pW1};
    const float* b1g[2] = {kb1, pb1};
    const float* b2g[2] = {kb2, pb2};
    const float* W3g[2] = {kW3, pW3};

    unsigned short* H1w = H1 + wave * 16 * WSTR;
    unsigned short* D2w = D2 + wave * 16 * WSTR;

    float qs = q0, ps = p0, accq = 0.f, accp = 0.f;
    const float dt = 0.05f;
    const float cc[3] = {0.5f, 0.5f, 1.0f};
    const float ww[4] = {1.f, 2.f, 2.f, 1.f};

#pragma unroll 1
    for (int s = 0; s < 4; ++s) {
        if (owner) {
            int mi = wave * 16 + lw;
            x0a[mi] = __sinf(qs);
            x1a[mi] = __cosf(qs);
            x2a[mi] = ps;
            gqa[mi] = 0.f;
            gpa[mi] = 0.f;
        }
#pragma unroll 1
        for (int br = 0; br < 2; ++br) {
            __syncthreads();                       // weight buffer reuse fence
            // ---- stage branch weights into LDS (f32 -> bf16) ----
            {
                const float* G = W2g[br];
                __builtin_prefetch(W2g[br ^ 1] + tid, 0, 1);   // warm next matrix
#pragma unroll 4
                for (int i = 0; i < HID; ++i)
                    Wl[i * WSTR + tid] = f2bf(G[i * HID + tid]);
                int c = (br == 0) ? 3 : 2;
                const float* g1 = W1g[br];
                float v0 = g1[tid * c + 0];
                float v1 = g1[tid * c + 1];
                float v2 = (br == 0) ? g1[tid * 3 + 2] : 0.f;  // V-net: no p input
                w1s[tid] = make_float4(v0, v1, v2, b1g[br][tid]);
                w3s[tid] = W3g[br][tid];
                b2s[tid] = b2g[br][tid];
            }
            __syncthreads();

            // ---- layer 1 forward: h1 = softplus(x @ W1^T + b1) -> bf16 staging ----
            for (int m = 0; m < 16; ++m) {
                float xm0 = x0a[wave * 16 + m];
                float xm1 = x1a[wave * 16 + m];
                float xm2 = x2a[wave * 16 + m];
#pragma unroll
                for (int c = 0; c < 8; ++c) {
                    int nn = c * 32 + lane;
                    float4 wv = w1s[nn];
                    float pre = wv.x * xm0 + wv.y * xm1 + wv.z * xm2 + wv.w;
                    H1w[m * WSTR + nn] = f2bf(softplus_f(pre));
                }
            }
            __syncthreads();

            // ---- layer 2 forward (WMMA) + grad seed d2 = W3 * sigmoid(pre2) ----
#pragma unroll 1
            for (int nt = 0; nt < 16; ++nt) {
                float bias = b2s[nt * 16 + lw];
                f32x8 acc;
#pragma unroll
                for (int r = 0; r < 8; ++r) acc[r] = bias;
#pragma unroll
                for (int kt = 0; kt < 8; ++kt) {
                    bf16x16 a = load_act_frag(H1w, lw, hf, kt);
                    bf16x16 b = load_wfwd_frag(Wl, lw, hf, nt, kt);
                    acc = wmma_bf16(a, b, acc);
                }
                int nn   = nt * 16 + lw;
                float w3v = w3s[nn];
#pragma unroll
                for (int r = 0; r < 8; ++r) {
                    int m = r + hf * 8;
                    D2w[m * WSTR + nn] = f2bf(w3v * sigmoid_f(acc[r]));
                }
            }
            __syncthreads();

            // ---- backward (WMMA): dh1 = d2 @ W2; dpre1 = dh1*sigma(pre1);
            //      dx_c = sum_j dpre1[j]*W1[j][c]  (sigma(pre1) recomputed) ----
            float xr0[8], xr1[8], xr2[8];
#pragma unroll
            for (int r = 0; r < 8; ++r) {
                int m = r + hf * 8;
                xr0[r] = x0a[wave * 16 + m];
                xr1[r] = x1a[wave * 16 + m];
                xr2[r] = x2a[wave * 16 + m];
            }
            float px0[8], px1[8], px2[8];
#pragma unroll
            for (int r = 0; r < 8; ++r) { px0[r] = 0.f; px1[r] = 0.f; px2[r] = 0.f; }

#pragma unroll 1
            for (int jt = 0; jt < 16; ++jt) {
                f32x8 acc;
#pragma unroll
                for (int r = 0; r < 8; ++r) acc[r] = 0.f;
#pragma unroll
                for (int kt = 0; kt < 8; ++kt) {
                    bf16x16 a = load_act_frag(D2w, lw, hf, kt);
                    bf16x16 b = load_wbwd_frag(Wl, lw, hf, jt, kt);
                    acc = wmma_bf16(a, b, acc);
                }
                int j = jt * 16 + lw;
                float4 wv = w1s[j];
#pragma unroll
                for (int r = 0; r < 8; ++r) {
                    float pre1 = wv.x * xr0[r] + wv.y * xr1[r] + wv.z * xr2[r] + wv.w;
                    float dp1  = acc[r] * sigmoid_f(pre1);
                    px0[r] += dp1 * wv.x;
                    px1[r] += dp1 * wv.y;
                    px2[r] += dp1 * wv.z;
                }
            }
            // reduce over the 16-lane j group (lanes 0-15 and 16-31 separately)
#pragma unroll
            for (int r = 0; r < 8; ++r) {
#pragma unroll
                for (int msk = 1; msk < 16; msk <<= 1) {
                    px0[r] += __shfl_xor(px0[r], msk);
                    px1[r] += __shfl_xor(px1[r], msk);
                    px2[r] += __shfl_xor(px2[r], msk);
                }
            }
            if (lw == 0) {                       // lanes 0 (m=0..7) and 16 (m=8..15)
#pragma unroll
                for (int r = 0; r < 8; ++r) {
                    int m = r + hf * 8;
                    // dH/dq += dx0*cos(q) - dx1*sin(q);  dH/dp += dx2
                    gqa[wave * 16 + m] += px0[r] * xr1[r] - px1[r] * xr0[r];
                    gpa[wave * 16 + m] += px2[r];
                }
            }
            __syncthreads();
        } // branch

        // ---- stage combine: dz = gH @ M^T + [0, Bu]; RK4 bookkeeping ----
        if (owner) {
            int mi = wave * 16 + lw;
            float g0 = gqa[mi], g1 = gpa[mi];
            float dq = g0 * M00 + g1 * M01;
            float dp = g0 * M10 + g1 * M11 + Bu;
            accq += ww[s] * dq;
            accp += ww[s] * dp;
            if (s < 3) {
                qs = q0 + cc[s] * dt * dq;
                ps = p0 + cc[s] * dt * dp;
            }
        }
    } // stage

    if (owner && row < n) {
        out[2 * row]     = q0 + (dt / 6.0f) * accq;
        out[2 * row + 1] = p0 + (dt / 6.0f) * accp;
    }
}

extern "C" void kernel_launch(void* const* d_in, const int* in_sizes, int n_in,
                              void* d_out, int out_size, void* d_ws, size_t ws_size,
                              hipStream_t stream) {
    const float* q    = (const float*)d_in[0];
    const float* p    = (const float*)d_in[1];
    const float* u    = (const float*)d_in[2];
    const float* kW1  = (const float*)d_in[3];
    const float* kb1  = (const float*)d_in[4];
    const float* kW2  = (const float*)d_in[5];
    const float* kb2  = (const float*)d_in[6];
    const float* kW3  = (const float*)d_in[7];
    // d_in[8] = kb3 (gradient-irrelevant)
    const float* pW1  = (const float*)d_in[9];
    const float* pb1  = (const float*)d_in[10];
    const float* pW2  = (const float*)d_in[11];
    const float* pb2  = (const float*)d_in[12];
    const float* pW3  = (const float*)d_in[13];
    // d_in[14] = pb3 (gradient-irrelevant)
    const float* Amat = (const float*)d_in[15];
    const float* Lp   = (const float*)d_in[16];
    const float* bc   = (const float*)d_in[17];

    int n = in_sizes[0];

    (void)hipFuncSetAttribute((const void*)phgnn_rk4,
                              hipFuncAttributeMaxDynamicSharedMemorySize,
                              SMEM_BYTES);

    dim3 grid((n + 127) / 128);
    phgnn_rk4<<<grid, 256, SMEM_BYTES, stream>>>(q, p, u,
                                                 kW1, kb1, kW2, kb2, kW3,
                                                 pW1, pb1, pW2, pb2, pW3,
                                                 Amat, Lp, bc,
                                                 (float*)d_out, n);
}
